// Block_549755814371
// MI455X (gfx1250) — matrix operand order repaired and checked by reading, not compiled, not run on hardware
//
#include <hip/hip_runtime.h>

#define N_EMBD 1024
#define N_HEAD 16
#define HEAD   64
#define BB     4
#define TT     2048
#define ROWS   (BB * TT)      // 8192
#define FFN    (4 * N_EMBD)   // 4096

typedef __attribute__((ext_vector_type(16))) __bf16 v16bf;
typedef __attribute__((ext_vector_type(8)))  float  v8f;
typedef __attribute__((ext_vector_type(4)))  unsigned int u32x4;
typedef __attribute__((ext_vector_type(8)))  int i32x8;
typedef __attribute__((ext_vector_type(4)))  int i32x4;

union Frag16 { v16bf v; unsigned short us[16]; };

__device__ __forceinline__ unsigned short f2bf(float f) {
  unsigned u = __float_as_uint(f);
  u += 0x7FFFu + ((u >> 16) & 1u);   // round-to-nearest-even
  return (unsigned short)(u >> 16);
}

// LDS byte offset of a __shared__ object (LDS aperture keeps offset in addr[31:0])
__device__ __forceinline__ unsigned lds_off(const void* p) {
  return (unsigned)(size_t)p;
}

// ---------------------------------------------------------------- TDM helper
// 2D tile (16-bit elements) global -> LDS via Tensor Data Mover.
// tile0 = contiguous elements per row, tile1 = rows, stride = row stride (elems).
__device__ __forceinline__ void tdm_load_2d(unsigned ldsoff, const void* gptr,
                                            unsigned tile0, unsigned tile1,
                                            unsigned tdim0, unsigned tdim1,
                                            unsigned stride) {
  unsigned long long ga = (unsigned long long)(size_t)gptr;
  u32x4 g0;
  g0[0] = 1u;                                   // count=1, user descriptor
  g0[1] = ldsoff;                               // lds_addr
  g0[2] = (unsigned)ga;                         // global_addr[31:0]
  g0[3] = (unsigned)(ga >> 32) | (2u << 30);    // global_addr[56:32] | type=2
  i32x8 g1;
  g1[0] = 0x00010000;                           // workgroup_mask=0, data_size=2B
  g1[1] = (int)((tdim0 & 0xFFFFu) << 16);       // tensor_dim0[15:0]
  g1[2] = (int)((tdim0 >> 16) | ((tdim1 & 0xFFFFu) << 16));
  g1[3] = (int)((tdim1 >> 16) | (tile0 << 16)); // tile_dim0
  g1[4] = (int)tile1;                           // tile_dim1 (tile_dim2=0)
  g1[5] = (int)stride;                          // tensor_dim0_stride[31:0]
  g1[6] = 0;
  g1[7] = 0;
  i32x4 z4 = {0, 0, 0, 0};
  i32x8 z8 = {0, 0, 0, 0, 0, 0, 0, 0};
  __builtin_amdgcn_tensor_load_to_lds(g0, g1, z4, z4, z8, 0);
}

// ---------------------------------------------------------------- converters
__global__ void cvt_f32_bf16(const float* __restrict__ in,
                             unsigned short* __restrict__ out, int n) {
  int i = blockIdx.x * 256 + threadIdx.x;
  if (i < n) out[i] = f2bf(in[i]);
}

// wq/wk/wv: (H, C, 64) fp32  ->  Wqkv bf16 [C][3*C], col = s*1024 + h*64 + d
__global__ void pack_qkv_w(const float* __restrict__ wq, const float* __restrict__ wk,
                           const float* __restrict__ wv, unsigned short* __restrict__ out) {
  int i = blockIdx.x * 256 + threadIdx.x;
  if (i >= N_EMBD * 3 * N_EMBD) return;
  int c = i / (3 * N_EMBD), col = i % (3 * N_EMBD);
  int s = col >> 10, rest = col & 1023;
  int h = rest >> 6, d = rest & 63;
  const float* w = (s == 0) ? wq : (s == 1) ? wk : wv;
  out[i] = f2bf(w[((size_t)h * N_EMBD + c) * HEAD + d]);
}

// ---------------------------------------------------------------- layernorm
__global__ __launch_bounds__(256) void ln_bf16(const float* __restrict__ x,
                                               const float* __restrict__ sc,
                                               const float* __restrict__ bi,
                                               unsigned short* __restrict__ out) {
  int row = blockIdx.x;
  int tid = threadIdx.x;
  float4 v = reinterpret_cast<const float4*>(x + (size_t)row * N_EMBD)[tid];
  float s  = v.x + v.y + v.z + v.w;
  float s2 = v.x * v.x + v.y * v.y + v.z * v.z + v.w * v.w;
  #pragma unroll
  for (int off = 16; off > 0; off >>= 1) {
    s  += __shfl_xor(s,  off, 32);
    s2 += __shfl_xor(s2, off, 32);
  }
  __shared__ float sh[16];
  int w = tid >> 5, ln = tid & 31;
  if (ln == 0) { sh[w] = s; sh[8 + w] = s2; }
  __syncthreads();
  if (tid == 0) {
    float a = 0.f, b = 0.f;
    for (int i = 0; i < 8; ++i) { a += sh[i]; b += sh[8 + i]; }
    sh[0] = a; sh[1] = b;
  }
  __syncthreads();
  float mu  = sh[0] * (1.0f / N_EMBD);
  float var = sh[1] * (1.0f / N_EMBD) - mu * mu;
  float rv  = rsqrtf(var + 1e-6f);
  int base = tid * 4;
  float e[4] = {v.x, v.y, v.z, v.w};
  #pragma unroll
  for (int i = 0; i < 4; ++i)
    out[(size_t)row * N_EMBD + base + i] =
        f2bf((e[i] - mu) * rv * sc[base + i] + bi[base + i]);
}

// ---------------------------------------------------------------- bf16 GEMM
// C[M,N] = A[M,K] * B[K,N]  (bf16 in, fp32 accumulate via WMMA)
// Block tile 128x128, 8 waves -> 16 rows x 128 cols each (8x v8f acc).
// K staged 32 deep via TDM double-buffered LDS tiles.
template <bool RELU, bool OUT_BF16>
__global__ __launch_bounds__(256) void gemm_ws(const unsigned short* __restrict__ A,
                                               const unsigned short* __restrict__ Bw,
                                               const float* __restrict__ bias,
                                               const float* __restrict__ res,
                                               void* __restrict__ outp,
                                               int M, int N, int K) {
  __shared__ __align__(16) unsigned short As[2][128 * 32];
  __shared__ __align__(16) unsigned short Bs[2][32 * 128];
  const int tid  = threadIdx.x;
  const int wave = tid >> 5, lane = tid & 31;
  const int hf = lane >> 4, ln = lane & 15;
  const int rowBase = blockIdx.y * 128, colBase = blockIdx.x * 128;
  v8f acc[8];
  v8f zero = {};
  #pragma unroll
  for (int nt = 0; nt < 8; ++nt) acc[nt] = zero;

  const unsigned short* Abase = A + (size_t)rowBase * K;
  const unsigned short* Bbase = Bw + colBase;

  if (tid == 0) {
    tdm_load_2d(lds_off(&As[0][0]), Abase, 32, 128, (unsigned)K, (unsigned)M, (unsigned)K);
    tdm_load_2d(lds_off(&Bs[0][0]), Bbase, 128, 32, (unsigned)N, (unsigned)K, (unsigned)N);
  }

  int buf = 0;
  for (int k0 = 0; k0 < K; k0 += 32) {
    if (tid == 0) {
      if (k0 + 32 < K) {   // prefetch next K-slab into the other buffer
        tdm_load_2d(lds_off(&As[buf ^ 1][0]), Abase + (k0 + 32), 32, 128,
                    (unsigned)K, (unsigned)M, (unsigned)K);
        tdm_load_2d(lds_off(&Bs[buf ^ 1][0]), Bbase + (size_t)(k0 + 32) * N, 128, 32,
                    (unsigned)N, (unsigned)K, (unsigned)N);
        __builtin_amdgcn_s_wait_tensorcnt((short)2);  // current buffer's pair done
      } else {
        __builtin_amdgcn_s_wait_tensorcnt((short)0);
      }
    }
    __syncthreads();

    Frag16 fa;
    #pragma unroll
    for (int v = 0; v < 8; ++v) {
      int kk = 2 * v + 8 * hf + ((v >= 4) ? 8 : 0);
      fa.us[2 * v]     = As[buf][(wave * 16 + ln) * 32 + kk];
      fa.us[2 * v + 1] = As[buf][(wave * 16 + ln) * 32 + kk + 1];
    }
    #pragma unroll
    for (int nt = 0; nt < 8; ++nt) {
      Frag16 fb;
      #pragma unroll
      for (int v = 0; v < 8; ++v) {
        int kk = 2 * v + 8 * hf + ((v >= 4) ? 8 : 0);
        fb.us[2 * v]     = Bs[buf][kk * 128 + nt * 16 + ln];
        fb.us[2 * v + 1] = Bs[buf][(kk + 1) * 128 + nt * 16 + ln];
      }
      acc[nt] = __builtin_amdgcn_wmma_f32_16x16x32_bf16(
          false, fa.v, false, fb.v, (short)0, acc[nt], false, false);
    }
    __syncthreads();
    buf ^= 1;
  }

  #pragma unroll
  for (int nt = 0; nt < 8; ++nt) {
    #pragma unroll
    for (int r = 0; r < 8; ++r) {
      int gr = rowBase + wave * 16 + r + 8 * hf;
      int gc = colBase + nt * 16 + ln;
      float val = acc[nt][r];
      if (bias) val += bias[gc];
      if (res)  val += res[(size_t)gr * N + gc];
      if (RELU) val = fmaxf(val, 0.0f);
      size_t o = (size_t)gr * N + gc;
      if (OUT_BF16) reinterpret_cast<unsigned short*>(outp)[o] = f2bf(val);
      else          reinterpret_cast<float*>(outp)[o] = val;
    }
  }
}

// ---------------------------------------------------------------- flash attention
// qkv: bf16 [B*T][3072] (cols: 0..1023 Q, 1024..2047 K, 2048..3071 V; head-major)
// O:   bf16 [B*T][1024]. Block = 128 query rows (8 waves x 16), causal, online softmax.
// K/V 32x64 tiles double-buffered in LDS via TDM.
__global__ __launch_bounds__(256) void flash_attn(const unsigned short* __restrict__ qkv,
                                                  unsigned short* __restrict__ O) {
  __shared__ __align__(16) unsigned short Kt[2][32 * 64];
  __shared__ __align__(16) unsigned short Vt[2][32 * 64];
  __shared__ unsigned short Ps[8][16 * 32];
  const int tid  = threadIdx.x;
  const int wave = tid >> 5, lane = tid & 31;
  const int hf = lane >> 4, ln = lane & 15;
  const int h = blockIdx.y, b = blockIdx.z;
  const int qBase = blockIdx.x * 128;
  const int qRow0 = qBase + wave * 16;
  const size_t rowStride = 3 * N_EMBD;
  const size_t batchBase = (size_t)b * TT;
  const float SCALE = 0.03125f;            // 1024^-0.5 (reference quirk)
  const unsigned short* kvBase = qkv + batchBase * rowStride + (size_t)h * HEAD;

  // Q fragments: 16x64 as two 16x32 A-frags (loaded once)
  Frag16 aq0, aq1;
  {
    const size_t qoff = (batchBase + qRow0 + ln) * rowStride + (size_t)h * HEAD;
    #pragma unroll
    for (int v = 0; v < 8; ++v) {
      int kk = 2 * v + 8 * hf + ((v >= 4) ? 8 : 0);
      aq0.us[2 * v]     = qkv[qoff + kk];
      aq0.us[2 * v + 1] = qkv[qoff + kk + 1];
      aq1.us[2 * v]     = qkv[qoff + 32 + kk];
      aq1.us[2 * v + 1] = qkv[qoff + 32 + kk + 1];
    }
  }

  float mx[8], sm[8], alpha[8];
  v8f o0 = {}, o1 = {}, o2 = {}, o3 = {};
  #pragma unroll
  for (int r = 0; r < 8; ++r) { mx[r] = -__builtin_inff(); sm[r] = 0.0f; }
  const int waveLast = qRow0 + 15;
  const int jEnd = qBase + 128;

  if (tid == 0) {
    tdm_load_2d(lds_off(&Kt[0][0]), kvBase + N_EMBD,     64, 32, 3072, 1u << 20, 3072);
    tdm_load_2d(lds_off(&Vt[0][0]), kvBase + 2 * N_EMBD, 64, 32, 3072, 1u << 20, 3072);
  }

  int buf = 0;
  for (int j = 0; j < jEnd; j += 32) {
    if (tid == 0) {
      if (j + 32 < jEnd) {
        size_t nb = (size_t)(j + 32) * rowStride;
        tdm_load_2d(lds_off(&Kt[buf ^ 1][0]), kvBase + nb + N_EMBD,     64, 32, 3072, 1u << 20, 3072);
        tdm_load_2d(lds_off(&Vt[buf ^ 1][0]), kvBase + nb + 2 * N_EMBD, 64, 32, 3072, 1u << 20, 3072);
        __builtin_amdgcn_s_wait_tensorcnt((short)2);
      } else {
        __builtin_amdgcn_s_wait_tensorcnt((short)0);
      }
    }
    __syncthreads();

    if (j <= waveLast) {   // wave-uniform causal skip
      // S = Q * K^T : two 16x16 score tiles, head-dim split into 2 x 32
      v8f s0 = {}, s1 = {};
      #pragma unroll
      for (int kt = 0; kt < 2; ++kt) {
        #pragma unroll
        for (int dc = 0; dc < 2; ++dc) {
          Frag16 fb;
          #pragma unroll
          for (int v = 0; v < 8; ++v) {
            int dd = dc * 32 + 2 * v + 8 * hf + ((v >= 4) ? 8 : 0);
            fb.us[2 * v]     = Kt[buf][(kt * 16 + ln) * 64 + dd];
            fb.us[2 * v + 1] = Kt[buf][(kt * 16 + ln) * 64 + dd + 1];
          }
          if (kt == 0)
            s0 = __builtin_amdgcn_wmma_f32_16x16x32_bf16(
                false, (dc == 0 ? aq0.v : aq1.v), false, fb.v, (short)0, s0, false, false);
          else
            s1 = __builtin_amdgcn_wmma_f32_16x16x32_bf16(
                false, (dc == 0 ? aq0.v : aq1.v), false, fb.v, (short)0, s1, false, false);
        }
      }
      // online softmax (rows live across 16 lanes of a half-wave)
      #pragma unroll
      for (int r = 0; r < 8; ++r) {
        int row = qRow0 + r + 8 * hf;
        float x0 = (j + ln      <= row) ? s0[r] * SCALE : -__builtin_inff();
        float x1 = (j + 16 + ln <= row) ? s1[r] * SCALE : -__builtin_inff();
        float mt = fmaxf(x0, x1);
        #pragma unroll
        for (int off = 8; off > 0; off >>= 1) mt = fmaxf(mt, __shfl_xor(mt, off, 32));
        float mn = fmaxf(mx[r], mt);
        float al = __expf(mx[r] - mn);
        float p0 = __expf(x0 - mn);
        float p1 = __expf(x1 - mn);
        float ps = p0 + p1;
        #pragma unroll
        for (int off = 8; off > 0; off >>= 1) ps += __shfl_xor(ps, off, 32);
        sm[r] = sm[r] * al + ps;
        mx[r] = mn;
        alpha[r] = al;
        Ps[wave][(r + 8 * hf) * 32 + ln]      = f2bf(p0);
        Ps[wave][(r + 8 * hf) * 32 + 16 + ln] = f2bf(p1);
      }
      #pragma unroll
      for (int r = 0; r < 8; ++r) {
        o0[r] *= alpha[r]; o1[r] *= alpha[r]; o2[r] *= alpha[r]; o3[r] *= alpha[r];
      }
      // P in A-layout (transposed through per-wave LDS scratch)
      Frag16 fp;
      #pragma unroll
      for (int v = 0; v < 8; ++v) {
        int kk = 2 * v + 8 * hf + ((v >= 4) ? 8 : 0);
        fp.us[2 * v]     = Ps[wave][ln * 32 + kk];
        fp.us[2 * v + 1] = Ps[wave][ln * 32 + kk + 1];
      }
      // O += P * V
      #pragma unroll
      for (int nt = 0; nt < 4; ++nt) {
        Frag16 fv;
        #pragma unroll
        for (int v = 0; v < 8; ++v) {
          int kk = 2 * v + 8 * hf + ((v >= 4) ? 8 : 0);
          fv.us[2 * v]     = Vt[buf][kk * 64 + nt * 16 + ln];
          fv.us[2 * v + 1] = Vt[buf][(kk + 1) * 64 + nt * 16 + ln];
        }
        v8f& oo = (nt == 0) ? o0 : (nt == 1) ? o1 : (nt == 2) ? o2 : o3;
        oo = __builtin_amdgcn_wmma_f32_16x16x32_bf16(
            false, fp.v, false, fv.v, (short)0, oo, false, false);
      }
    }
    __syncthreads();
    buf ^= 1;
  }

  float rs[8];
  #pragma unroll
  for (int r = 0; r < 8; ++r) rs[r] = 1.0f / sm[r];
  v8f oarr[4] = {o0, o1, o2, o3};
  #pragma unroll
  for (int nt = 0; nt < 4; ++nt) {
    #pragma unroll
    for (int r = 0; r < 8; ++r) {
      int t = qRow0 + r + 8 * hf;
      O[(batchBase + t) * N_EMBD + h * HEAD + nt * 16 + ln] = f2bf(oarr[nt][r] * rs[r]);
    }
  }
}

// ---------------------------------------------------------------- launch
extern "C" void kernel_launch(void* const* d_in, const int* in_sizes, int n_in,
                              void* d_out, int out_size, void* d_ws, size_t ws_size,
                              hipStream_t stream) {
  (void)in_sizes; (void)n_in; (void)out_size; (void)ws_size;
  const float* x   = (const float*)d_in[0];
  const float* wq  = (const float*)d_in[1];
  const float* wk  = (const float*)d_in[2];
  const float* wv  = (const float*)d_in[3];
  const float* wp  = (const float*)d_in[4];
  const float* bp  = (const float*)d_in[5];
  const float* l1s = (const float*)d_in[6];
  const float* l1b = (const float*)d_in[7];
  const float* l2s = (const float*)d_in[8];
  const float* l2b = (const float*)d_in[9];
  const float* w1  = (const float*)d_in[10];
  const float* b1  = (const float*)d_in[11];
  const float* w2  = (const float*)d_in[12];
  const float* b2  = (const float*)d_in[13];
  float* out = (float*)d_out;

  char* ws = (char*)d_ws;
  size_t off = 0;
  auto alloc = [&](size_t bytes) -> void* {
    void* p = ws + off;
    off += (bytes + 255) & ~(size_t)255;
    return p;
  };
  unsigned short* h_bf = (unsigned short*)alloc((size_t)ROWS * N_EMBD * 2);
  unsigned short* wqkv = (unsigned short*)alloc((size_t)N_EMBD * 3 * N_EMBD * 2);
  unsigned short* qkv  = (unsigned short*)alloc((size_t)ROWS * 3 * N_EMBD * 2);
  unsigned short* Obf  = (unsigned short*)alloc((size_t)ROWS * N_EMBD * 2);
  unsigned short* wpb  = (unsigned short*)alloc((size_t)N_EMBD * N_EMBD * 2);
  float*          x1   = (float*)alloc((size_t)ROWS * N_EMBD * 4);
  unsigned short* h2   = (unsigned short*)alloc((size_t)ROWS * N_EMBD * 2);
  unsigned short* w1b  = (unsigned short*)alloc((size_t)N_EMBD * FFN * 2);
  unsigned short* actb = (unsigned short*)alloc((size_t)ROWS * FFN * 2);
  unsigned short* w2b  = (unsigned short*)alloc((size_t)FFN * N_EMBD * 2);

  // weight repack (fp32 -> bf16)
  pack_qkv_w<<<(N_EMBD * 3 * N_EMBD) / 256, 256, 0, stream>>>(wq, wk, wv, wqkv);
  cvt_f32_bf16<<<(N_EMBD * N_EMBD) / 256, 256, 0, stream>>>(wp, wpb, N_EMBD * N_EMBD);
  cvt_f32_bf16<<<(N_EMBD * FFN) / 256, 256, 0, stream>>>(w1, w1b, N_EMBD * FFN);
  cvt_f32_bf16<<<(FFN * N_EMBD) / 256, 256, 0, stream>>>(w2, w2b, FFN * N_EMBD);

  // h = LN1(x)  (bf16)
  ln_bf16<<<ROWS, 256, 0, stream>>>(x, l1s, l1b, h_bf);
  // qkv = h @ Wqkv   (8192 x 3072 x 1024)
  gemm_ws<false, true><<<dim3(3 * N_EMBD / 128, ROWS / 128), 256, 0, stream>>>(
      h_bf, wqkv, nullptr, nullptr, qkv, ROWS, 3 * N_EMBD, N_EMBD);
  // attention (flash, causal)
  flash_attn<<<dim3(TT / 128, N_HEAD, BB), 256, 0, stream>>>(qkv, Obf);
  // x1 = x + O @ Wproj + b_proj   (fp32)
  gemm_ws<false, false><<<dim3(N_EMBD / 128, ROWS / 128), 256, 0, stream>>>(
      Obf, wpb, bp, x, x1, ROWS, N_EMBD, N_EMBD);
  // h2 = LN2(x1) (bf16)
  ln_bf16<<<ROWS, 256, 0, stream>>>(x1, l2s, l2b, h2);
  // act = relu(h2 @ W1 + b1)  (bf16)
  gemm_ws<true, true><<<dim3(FFN / 128, ROWS / 128), 256, 0, stream>>>(
      h2, w1b, b1, nullptr, actb, ROWS, FFN, N_EMBD);
  // out = x1 + act @ W2 + b2  (fp32)
  gemm_ws<false, false><<<dim3(N_EMBD / 128, ROWS / 128), 256, 0, stream>>>(
      actb, w2b, b2, x1, out, ROWS, N_EMBD, FFN);
}
